// GPU_NTM_26190710571537
// MI455X (gfx1250) — matrix operand order repaired and checked by reading, not compiled, not run on hardware
//
#include <hip/hip_runtime.h>

// ---------------------------------------------------------------------------
// NTM/GRU recurrent network on MI455X (gfx1250).
//  * bf16 WMMA (v_wmma_f32_16x16x32_bf16) for every GEMM, f32 accumulate
//  * persistent kernel: 64 blocks = 4 batch-groups (16 rows = 1 WMMA M tile)
//    x 16 H-slices; 384 threads = 12 wave32s, one 16x16 tile per wave
//  * weights converted to bf16 once per launch -> L2-resident (~10.5MB)
//  * grid-wide sense-reversing barrier (2 per timestep) for the recurrence
// ---------------------------------------------------------------------------

typedef __bf16 bf16x16 __attribute__((ext_vector_type(16)));
typedef __bf16 bf16x8  __attribute__((ext_vector_type(8)));
typedef float  f32x8   __attribute__((ext_vector_type(8)));
typedef float  f32x4   __attribute__((ext_vector_type(4)));

#define BB 64
#define TT 512
#define II 512
#define HH 1024
#define OO 512
#define PP 1024   // K*N memory slots
#define MM 64     // memory word width
#define H3 3072

#define NBLK  64
#define WAVES 12
#define BLOCK 384

// ---------------------------------------------------------------------------
// Prep: f32 -> bf16 weight conversion, mem transpose, zero h / barrier.
// ---------------------------------------------------------------------------
__global__ void ntm_prep(const float* __restrict__ Wk,  const float* __restrict__ Wri,
                         const float* __restrict__ Wih, const float* __restrict__ Whh,
                         const float* __restrict__ Wout,const float* __restrict__ mem,
                         __bf16* Wk_bf, __bf16* Wri_bf, __bf16* Wih_bf, __bf16* Whh_bf,
                         __bf16* Wout_bf, __bf16* mem_bf, __bf16* memT_bf,
                         float* h_f32, __bf16* h_bf, unsigned* bar)
{
    size_t i0 = (size_t)blockIdx.x * blockDim.x + threadIdx.x;
    size_t st = (size_t)gridDim.x * blockDim.x;
    for (size_t i = i0; i < (size_t)H3 * HH; i += st) Whh_bf[i]  = (__bf16)Whh[i];
    for (size_t i = i0; i < (size_t)H3 * II; i += st) Wih_bf[i]  = (__bf16)Wih[i];
    for (size_t i = i0; i < (size_t)OO * HH; i += st) Wout_bf[i] = (__bf16)Wout[i];
    for (size_t i = i0; i < (size_t)MM * HH; i += st) Wk_bf[i]   = (__bf16)Wk[i];
    for (size_t i = i0; i < (size_t)II * MM; i += st) Wri_bf[i]  = (__bf16)Wri[i];
    for (size_t i = i0; i < (size_t)PP * MM; i += st) {
        float v = mem[i];
        mem_bf[i] = (__bf16)v;                    // (p, m) row-major
        size_t p = i >> 6, m = i & 63;
        memT_bf[m * (size_t)PP + p] = (__bf16)v;  // (m, p) row-major
    }
    for (size_t i = i0; i < (size_t)BB * HH; i += st) { h_f32[i] = 0.f; h_bf[i] = (__bf16)0.f; }
    if (i0 < 2) bar[i0] = 0u;
}

// ---------------------------------------------------------------------------
// Persistent recurrent kernel.
// ---------------------------------------------------------------------------
__global__ __launch_bounds__(BLOCK) void ntm_persistent(
    const float* __restrict__ x,
    const float* __restrict__ bk,  const float* __restrict__ bri,
    const float* __restrict__ bih, const float* __restrict__ bhh,
    const float* __restrict__ bout,
    const __bf16* __restrict__ Whh_bf, const __bf16* __restrict__ Wih_bf,
    const __bf16* __restrict__ Wout_bf, const __bf16* __restrict__ Wk_bf,
    const __bf16* __restrict__ Wri_bf, const __bf16* __restrict__ mem_bf,
    const __bf16* __restrict__ memT_bf,
    float* __restrict__ h_f32, __bf16* __restrict__ h_bf,
    __bf16* __restrict__ xin_bf, unsigned* bar, float* __restrict__ out)
{
    const int role = blockIdx.x & 15;          // H-slice [64*role, 64*role+64)
    const int b0   = (blockIdx.x >> 4) * 16;   // batch rows [b0, b0+16)
    const int tid  = threadIdx.x;
    const int wave = tid >> 5;
    const int lane = tid & 31;
    const int arow = lane & 15;
    const int hi   = (lane & 16) ? 1 : 0;
    const int ahalf = hi ? 8  : 0;             // A operand K sub-block offset
    const int bhalf = hi ? 8  : 0;             // B operand K sub-block offset
    const int r0    = hi ? 8  : 0;             // C/D row base for this half-wave

    __shared__ float gh_lds[16 * 192];                         // persists CA->B
    __shared__ __align__(16) char smem[16*64*4 + 16*1024*2 + 16*64*4];
    float*  gi_lds = (float*)smem;                             // 16x192 (phase B)
    float*  kq_lds = (float*)smem;                             // 16x64  (read path)
    __bf16* sc_lds = (__bf16*)(smem + 16*64*4);                // 16x1024 scores/softmax
    float*  rd_lds = (float*)(smem + 16*64*4 + 16*1024*2);     // 16x64  read vector

    // D = A(16x32 bf16) x B(32x16 bf16) + C, K-loop; A,B row-major (n-major for B)
    auto gemm_bf = [&](const __bf16* __restrict__ A, int lda,
                       const __bf16* __restrict__ Bw, int ldb,
                       int n0, int K) -> f32x8 {
        f32x8 acc = {};
        const __bf16* pa = A + (size_t)arow * lda + ahalf;
        const __bf16* pb = Bw + (size_t)(n0 + arow) * ldb + bhalf;
        for (int k = 0; k < K; k += 32) {
            bf16x8 alo = *(const bf16x8*)(pa + k);
            bf16x8 ah2 = *(const bf16x8*)(pa + k + 16);
            bf16x16 a, b;
            { bf16x8 blo = *(const bf16x8*)(pb + k); bf16x8 bh2 = *(const bf16x8*)(pb + k + 16);
              for (int i = 0; i < 8; ++i) { b[i] = blo[i]; b[i + 8] = bh2[i]; } }
            #pragma unroll
            for (int i = 0; i < 8; ++i) { a[i] = alo[i]; a[i + 8] = ah2[i]; }
            __builtin_prefetch((const void*)(pb + k + 64), 0, 1); // global_prefetch_b8
            acc = __builtin_amdgcn_wmma_f32_16x16x32_bf16(false, a, false, b,
                                                          (short)0, acc, false, false);
        }
        return acc;
    };

    // Same, but A sourced from f32 (LDS) with on-the-fly bf16 convert.
    auto gemm_f32a = [&](const float* __restrict__ Af, int lda,
                         const __bf16* __restrict__ Bw, int ldb,
                         int n0, int K) -> f32x8 {
        f32x8 acc = {};
        const float* pa = Af + (size_t)arow * lda + ahalf;
        const __bf16* pb = Bw + (size_t)(n0 + arow) * ldb + bhalf;
        for (int k = 0; k < K; k += 32) {
            f32x4 a0 = *(const f32x4*)(pa + k);
            f32x4 a1 = *(const f32x4*)(pa + k + 4);
            f32x4 a2 = *(const f32x4*)(pa + k + 16);
            f32x4 a3 = *(const f32x4*)(pa + k + 20);
            bf16x16 a;
            #pragma unroll
            for (int i = 0; i < 4; ++i) {
                a[i]      = (__bf16)a0[i]; a[i + 4]  = (__bf16)a1[i];
                a[i + 8]  = (__bf16)a2[i]; a[i + 12] = (__bf16)a3[i];
            }
            bf16x16 b;
            { bf16x8 blo = *(const bf16x8*)(pb + k); bf16x8 bh2 = *(const bf16x8*)(pb + k + 16);
              for (int i = 0; i < 8; ++i) { b[i] = blo[i]; b[i + 8] = bh2[i]; } }
            acc = __builtin_amdgcn_wmma_f32_16x16x32_bf16(false, a, false, b,
                                                          (short)0, acc, false, false);
        }
        return acc;
    };

    auto gridbar = [&]() {
        __syncthreads();
        if (tid == 0) {
            __threadfence();
            unsigned g = __atomic_load_n(&bar[1], __ATOMIC_RELAXED);
            if (atomicAdd(&bar[0], 1u) == NBLK - 1u) {
                __atomic_store_n(&bar[0], 0u, __ATOMIC_RELAXED);
                __threadfence();
                __atomic_store_n(&bar[1], g + 1u, __ATOMIC_RELEASE);
            } else {
                while (__atomic_load_n(&bar[1], __ATOMIC_ACQUIRE) == g)
                    __builtin_amdgcn_s_sleep(1);
            }
        }
        __syncthreads();
    };

    // Phase B: gi = xin @ Wih^T (own 192 cols), GRU gate math, update h slice.
    auto phaseB = [&]() {
        {
            int g = wave >> 2, j = wave & 3;
            int n0 = g * HH + role * 64 + j * 16;
            f32x8 acc = gemm_bf(xin_bf + (size_t)b0 * II, II, Wih_bf, II, n0, II);
            int col = g * 64 + j * 16 + arow;
            float bias = bih[n0 + arow];
            #pragma unroll
            for (int v = 0; v < 8; ++v) gi_lds[(r0 + v) * 192 + col] = acc[v] + bias;
        }
        __syncthreads();
        for (int idx = tid; idx < 16 * 64; idx += BLOCK) {
            int m = idx >> 6, c = idx & 63;
            float ir = gi_lds[m * 192 + c];
            float iz = gi_lds[m * 192 + 64 + c];
            float in_ = gi_lds[m * 192 + 128 + c];
            float hr = gh_lds[m * 192 + c];
            float hz = gh_lds[m * 192 + 64 + c];
            float hn = gh_lds[m * 192 + 128 + c];
            size_t hidx = (size_t)(b0 + m) * HH + role * 64 + c;
            float hold = h_f32[hidx];
            float rg = 1.f / (1.f + __expf(-(ir + hr)));
            float zg = 1.f / (1.f + __expf(-(iz + hz)));
            float ng = tanhf(in_ + rg * hn);
            float hnew = (1.f - zg) * ng + zg * hold;
            h_f32[hidx] = hnew;
            h_bf[hidx]  = (__bf16)hnew;
        }
    };

    // Phase CA: out(t) from full h; then gh and EMM read path for t+1.
    auto phaseCA = [&](int t, bool do_out, bool has_next) {
        if (do_out && wave < 2) {                          // out slice: 16x32
            int n0 = role * 32 + wave * 16;
            f32x8 acc = gemm_bf(h_bf + (size_t)b0 * HH, HH, Wout_bf, HH, n0, HH);
            int col = n0 + arow;
            float bias = bout[col];
            #pragma unroll
            for (int v = 0; v < 8; ++v) {
                float val = 1.f / (1.f + __expf(-(acc[v] + bias)));
                out[((size_t)(b0 + r0 + v) * TT + t) * OO + col] = val;
            }
        }
        if (role == 0 && has_next) {                       // EMM read path (t+1)
            if (wave < 4) {                                // kq = h@Wk^T + bk : 16x64
                int n0 = wave * 16;
                f32x8 acc = gemm_bf(h_bf + (size_t)b0 * HH, HH, Wk_bf, HH, n0, HH);
                int col = n0 + arow;
                float bias = bk[col];
                #pragma unroll
                for (int v = 0; v < 8; ++v) kq_lds[(r0 + v) * 64 + col] = acc[v] + bias;
            }
            __syncthreads();
            for (int tp = wave; tp < 64; tp += WAVES) {    // scores = kq@mem^T : 16x1024
                int p0 = tp * 16;
                f32x8 acc = gemm_f32a(kq_lds, 64, mem_bf, MM, p0, MM);
                int col = p0 + arow;
                #pragma unroll
                for (int v = 0; v < 8; ++v) sc_lds[(r0 + v) * PP + col] = (__bf16)acc[v];
            }
            __syncthreads();
            for (int r = wave; r < 16; r += WAVES) {       // row softmax (wave32 shfl)
                float vals[32];
                float mx = -3.0e38f;
                #pragma unroll
                for (int i = 0; i < 32; ++i) {
                    float v = (float)sc_lds[r * PP + lane + i * 32];
                    vals[i] = v; mx = fmaxf(mx, v);
                }
                #pragma unroll
                for (int s = 16; s > 0; s >>= 1) mx = fmaxf(mx, __shfl_xor(mx, s, 32));
                float sum = 0.f;
                #pragma unroll
                for (int i = 0; i < 32; ++i) { vals[i] = __expf(vals[i] - mx); sum += vals[i]; }
                #pragma unroll
                for (int s = 16; s > 0; s >>= 1) sum += __shfl_xor(sum, s, 32);
                float inv = 1.f / sum;
                #pragma unroll
                for (int i = 0; i < 32; ++i)
                    sc_lds[r * PP + lane + i * 32] = (__bf16)(vals[i] * inv);
            }
            __syncthreads();
            if (wave < 4) {                                // read = w@mem : 16x64, K=1024
                int n0 = wave * 16;
                f32x8 acc = gemm_bf(sc_lds, PP, memT_bf, PP, n0, PP);
                int col = n0 + arow;
                #pragma unroll
                for (int v = 0; v < 8; ++v) rd_lds[(r0 + v) * 64 + col] = acc[v];
            }
            __syncthreads();
            for (int tn = wave; tn < 32; tn += WAVES) {    // xin = x + relu(read@Wri^T+bri)
                int n0 = tn * 16;
                f32x8 acc = gemm_f32a(rd_lds, 64, Wri_bf, MM, n0, MM);
                int col = n0 + arow;
                float bias = bri[col];
                #pragma unroll
                for (int v = 0; v < 8; ++v) {
                    float rv = fmaxf(acc[v] + bias, 0.f);
                    float xv = x[((size_t)(b0 + r0 + v) * TT + (t + 1)) * II + col];
                    xin_bf[(size_t)(b0 + r0 + v) * II + col] = (__bf16)(xv + rv);
                }
            }
        }
        if (has_next) {                                    // gh = h@Whh^T : own 192 cols
            int g = wave >> 2, j = wave & 3;
            int n0 = g * HH + role * 64 + j * 16;
            f32x8 acc = gemm_bf(h_bf + (size_t)b0 * HH, HH, Whh_bf, HH, n0, HH);
            int col = g * 64 + j * 16 + arow;
            float bias = bhh[n0 + arow];
            #pragma unroll
            for (int v = 0; v < 8; ++v) gh_lds[(r0 + v) * 192 + col] = acc[v] + bias;
        }
    };

    phaseCA(-1, false, true);                              // prime gh/xin for t=0 (h=0)
    gridbar();
    for (int t = 0; t < TT; ++t) {
        phaseB();
        gridbar();
        phaseCA(t, true, t + 1 < TT);
        gridbar();
    }
}

// ---------------------------------------------------------------------------
extern "C" void kernel_launch(void* const* d_in, const int* in_sizes, int n_in,
                              void* d_out, int out_size, void* d_ws, size_t ws_size,
                              hipStream_t stream)
{
    (void)in_sizes; (void)n_in; (void)out_size; (void)ws_size;
    const float* x    = (const float*)d_in[0];
    const float* mem  = (const float*)d_in[1];
    const float* Wk   = (const float*)d_in[2];
    const float* bk   = (const float*)d_in[3];
    const float* Wri  = (const float*)d_in[4];
    const float* bri  = (const float*)d_in[5];
    const float* Wih  = (const float*)d_in[6];
    const float* Whh  = (const float*)d_in[7];
    const float* bih  = (const float*)d_in[8];
    const float* bhh  = (const float*)d_in[9];
    const float* Wout = (const float*)d_in[10];
    const float* bout = (const float*)d_in[11];
    float* out = (float*)d_out;

    char* ws = (char*)d_ws;
    size_t off = 0;
    auto take = [&](size_t bytes) -> char* {
        char* p = ws + off;
        off = (off + bytes + 255) & ~(size_t)255;
        return p;
    };
    __bf16* Whh_bf  = (__bf16*)take((size_t)H3 * HH * 2);
    __bf16* Wih_bf  = (__bf16*)take((size_t)H3 * II * 2);
    __bf16* Wout_bf = (__bf16*)take((size_t)OO * HH * 2);
    __bf16* Wk_bf   = (__bf16*)take((size_t)MM * HH * 2);
    __bf16* Wri_bf  = (__bf16*)take((size_t)II * MM * 2);
    __bf16* mem_bf  = (__bf16*)take((size_t)PP * MM * 2);
    __bf16* memT_bf = (__bf16*)take((size_t)MM * PP * 2);
    float*  h_f32   = (float*)take((size_t)BB * HH * 4);
    __bf16* h_bf    = (__bf16*)take((size_t)BB * HH * 2);
    __bf16* xin_bf  = (__bf16*)take((size_t)BB * II * 2);
    unsigned* bar   = (unsigned*)take(256);

    ntm_prep<<<1024, 256, 0, stream>>>(Wk, Wri, Wih, Whh, Wout, mem,
                                       Wk_bf, Wri_bf, Wih_bf, Whh_bf, Wout_bf,
                                       mem_bf, memT_bf, h_f32, h_bf, bar);

    ntm_persistent<<<NBLK, BLOCK, 0, stream>>>(x, bk, bri, bih, bhh, bout,
                                               Whh_bf, Wih_bf, Wout_bf, Wk_bf,
                                               Wri_bf, mem_bf, memT_bf,
                                               h_f32, h_bf, xin_bf, bar, out);
}
